// BertEncoder_45260365365543
// MI455X (gfx1250) — matrix operand order repaired and checked by reading, not compiled, not run on hardware
//
#include <hip/hip_runtime.h>
#include <hip/hip_bf16.h>
#include <math.h>

typedef __bf16 bf16_t;
typedef __attribute__((ext_vector_type(16))) __bf16 v16bf;
typedef __attribute__((ext_vector_type(8)))  __bf16 v8bf;
typedef __attribute__((ext_vector_type(8)))  float  v8f;

#define NL   12
#define H    768
#define NH   12
#define DH   64
#define FF   3072
#define SEQ  512
#define BB   4
#define NTOK (BB * SEQ)   // 2048
#define NOPS 16

#define KSTEP 64
#define LDP   72          // padded LDS row stride (halves): conflict-free frags

// Async global -> LDS copy of 16 bytes (CDNA5 GLOBAL_LOAD_ASYNC_TO_LDS_B128,
// tracked by ASYNCcnt). ldsoff = LDS byte offset (low 32 bits of flat shared
// address), g = per-lane global address.
__device__ __forceinline__ void async_b128(unsigned ldsoff, const bf16_t* g)
{
    asm volatile("global_load_async_to_lds_b128 %0, %1, off"
                 :: "v"(ldsoff), "v"(g)
                 : "memory");
}
__device__ __forceinline__ void wait_async6()
{
    asm volatile("s_wait_asynccnt 0x6" ::: "memory");
}
__device__ __forceinline__ void wait_async0()
{
    asm volatile("s_wait_asynccnt 0x0" ::: "memory");
}

// ---------------------------------------------------------------------------
// Generic bf16 TN GEMM:  C[M,N] = A[M,K] * Bt[N,K]^T  (+bias, +gelu)
// A, Bt row-major bf16. f32 accumulation via v_wmma_f32_16x16x32_bf16.
// 256 threads = 8 waves; block tile 64x128; wave tile 32x32 (2x2 WMMA);
// K-step 64 (8 WMMA per barrier interval). Double-buffered LDS fed by
// GLOBAL_LOAD_ASYNC_TO_LDS_B128 + s_wait_asynccnt.
// Requires: M % 64 == 0, K % 64 == 0. N handled with guards (Bt rows clamped
// so per-wave ASYNCcnt stays uniform; OOB columns discarded at store).
// Two-level batching: z = z1*nbatch2 + z2, strides (s?1, s?2) in elements.
// ---------------------------------------------------------------------------
__global__ __launch_bounds__(256) void gemm_bf16_tn(
    const bf16_t* __restrict__ A, int lda, long long sA1, long long sA2,
    const bf16_t* __restrict__ Bt, int ldb, long long sB1, long long sB2,
    float* __restrict__ Cf, bf16_t* __restrict__ Cb, int ldc,
    long long sC1, long long sC2,
    const float* __restrict__ bias, int M, int N, int K, int nbatch2, int doGelu)
{
    __shared__ bf16_t As[2][64 * LDP];    // 2 x (64 x 64) tile, stride 72
    __shared__ bf16_t Bs[2][128 * LDP];   // 2 x (128 x 64) tile (N-major)

    const int z  = blockIdx.z;
    const int z1 = z / nbatch2;
    const int z2 = z - z1 * nbatch2;
    const bf16_t* Ab = A  + z1 * sA1 + z2 * sA2;
    const bf16_t* Bb = Bt + z1 * sB1 + z2 * sB2;
    const long long cbase = z1 * sC1 + z2 * sC2;

    const int tid  = threadIdx.x;
    const int lane = tid & 31;
    const int wave = tid >> 5;
    const int wm   = wave >> 2;   // 0..1
    const int wn   = wave & 3;    // 0..3
    const int l    = lane & 15;
    const int hi   = lane >> 4;

    const int gm0 = blockIdx.y * 64;
    const int gn0 = blockIdx.x * 128;

    // cooperative async-load slices (per thread: A 2x16B, B 4x16B = 6 loads)
    const int ar = tid >> 2;            // 0..63   (A row in tile)
    const int ak = (tid & 3) * 16;      // 0,16,32,48
    const int br = tid >> 1;            // 0..127  (Bt row in tile)
    const int bk = (tid & 1) * 32;      // 0,32
    int gnB = gn0 + br;
    if (gnB >= N) gnB = N - 1;          // clamp: keeps ASYNCcnt uniform

    const bf16_t* agp = Ab + (long long)(gm0 + ar) * lda + ak;
    const bf16_t* bgp = Bb + (long long)gnB * ldb + bk;

    const unsigned aoff[2] = {
        (unsigned)(size_t)&As[0][ar * LDP + ak],
        (unsigned)(size_t)&As[1][ar * LDP + ak] };
    const unsigned boff[2] = {
        (unsigned)(size_t)&Bs[0][br * LDP + bk],
        (unsigned)(size_t)&Bs[1][br * LDP + bk] };

    auto issue = [&](int buf, int k0) {
        const bf16_t* ap = agp + k0;
        const bf16_t* bp = bgp + k0;
        const unsigned a0 = aoff[buf];
        const unsigned b0 = boff[buf];
        async_b128(a0,      ap);
        async_b128(a0 + 16, ap + 8);
        async_b128(b0,      bp);
        async_b128(b0 + 16, bp + 8);
        async_b128(b0 + 32, bp + 16);
        async_b128(b0 + 48, bp + 24);
    };

    v8f acc[2][2] = {};

    const int nk = K / KSTEP;
    issue(0, 0);

    for (int kt = 0; kt < nk; ++kt) {
        const int buf = kt & 1;
        if (kt + 1 < nk) {
            issue(buf ^ 1, (kt + 1) * KSTEP);
            wait_async6();              // 6 newest in flight; older tile done
        } else {
            wait_async0();
        }
        __syncthreads();

        const bf16_t* Asb = As[buf];
        const bf16_t* Bsb = Bs[buf];
        #pragma unroll
        for (int s = 0; s < 2; ++s) {
            const int ks = s * 32;
            // A fragment (16x32 bf16): lane l row M=l; hi=0 -> K 0..7 &
            // 16..23, hi=1 -> K 8..15 & 24..31 (ISA 7.12.2 16-bit A layout).
            union { v16bf v; v8bf h[2]; } fa[2], fb[2];
            #pragma unroll
            for (int i = 0; i < 2; ++i) {
                const bf16_t* ap = Asb + (wm * 32 + i * 16 + l) * LDP + ks;
                fa[i].h[0] = *(const v8bf*)(ap + hi * 8);
                fa[i].h[1] = *(const v8bf*)(ap + 16 + hi * 8);
            }
            // B fragment (32x16 bf16): lane l col N=l; hi=0 -> K 0..7 &
            // 16..23, hi=1 -> K 8..15 & 24..31 (same K order as A).
            #pragma unroll
            for (int j = 0; j < 2; ++j) {
                const bf16_t* bp = Bsb + (wn * 32 + j * 16 + l) * LDP + ks + hi * 8;
                fb[j].h[0] = *(const v8bf*)bp;
                fb[j].h[1] = *(const v8bf*)(bp + 16);
            }
            #pragma unroll
            for (int i = 0; i < 2; ++i)
                #pragma unroll
                for (int j = 0; j < 2; ++j)
                    acc[i][j] = __builtin_amdgcn_wmma_f32_16x16x32_bf16(
                        false, fa[i].v, false, fb[j].v, (short)0, acc[i][j],
                        false, false);
        }
        __syncthreads();
    }

    // C layout (ISA 7.12.2): lane col N = l; VGPR r -> row M = hi*8 + r.
    #pragma unroll
    for (int j = 0; j < 2; ++j) {
        const int col = gn0 + wn * 32 + j * 16 + l;
        if (col >= N) continue;
        const float bv = bias ? bias[col] : 0.0f;
        #pragma unroll
        for (int i = 0; i < 2; ++i) {
            const int row0 = gm0 + wm * 32 + i * 16 + hi * 8;
            #pragma unroll
            for (int r = 0; r < 8; ++r) {
                float v = acc[i][j][r] + bv;
                if (doGelu) v = 0.5f * v * (1.0f + erff(v * 0.70710678118f));
                const long long off = cbase + (long long)(row0 + r) * ldc + col;
                if (Cf) Cf[off] = v;
                if (Cb) Cb[off] = (bf16_t)v;
            }
        }
    }
}

// ---------------------------------------------------------------------------
// Embedding gather + LayerNorm -> x (f32) and xb (bf16). One block per token.
// ---------------------------------------------------------------------------
__global__ __launch_bounds__(256) void embed_ln(
    const long long* __restrict__ ids, const float* __restrict__ wemb,
    const float* __restrict__ pemb, const float* __restrict__ temb,
    const float* __restrict__ g, const float* __restrict__ bta,
    float* __restrict__ x, bf16_t* __restrict__ xb)
{
    const int tok = blockIdx.x;
    const int s   = tok & (SEQ - 1);
    const int tid = threadIdx.x;
    const long long id = ids[tok];
    __shared__ float red[256];

    float vals[3];
    float sum = 0.f;
    #pragma unroll
    for (int e = 0; e < 3; ++e) {
        const int h = e * 256 + tid;
        const float v = wemb[id * H + h] + pemb[(long long)s * H + h] + temb[h];
        vals[e] = v; sum += v;
    }
    red[tid] = sum; __syncthreads();
    for (int off = 128; off > 0; off >>= 1) {
        if (tid < off) red[tid] += red[tid + off];
        __syncthreads();
    }
    const float mean = red[0] * (1.0f / H);
    __syncthreads();
    float ss = 0.f;
    #pragma unroll
    for (int e = 0; e < 3; ++e) { const float d = vals[e] - mean; ss += d * d; }
    red[tid] = ss; __syncthreads();
    for (int off = 128; off > 0; off >>= 1) {
        if (tid < off) red[tid] += red[tid + off];
        __syncthreads();
    }
    const float rstd = rsqrtf(red[0] * (1.0f / H) + 1e-12f);
    #pragma unroll
    for (int e = 0; e < 3; ++e) {
        const int h = e * 256 + tid;
        const float v = (vals[e] - mean) * rstd * g[h] + bta[h];
        x[(long long)tok * H + h]  = v;
        xb[(long long)tok * H + h] = (bf16_t)v;
    }
}

// ---------------------------------------------------------------------------
// x = LayerNorm(x + y) -> x (f32, in place) and xb (bf16). One block per token.
// ---------------------------------------------------------------------------
__global__ __launch_bounds__(256) void ln_residual(
    float* __restrict__ x, const float* __restrict__ y,
    const float* __restrict__ g, const float* __restrict__ bta,
    bf16_t* __restrict__ xb)
{
    const int tok = blockIdx.x;
    const int tid = threadIdx.x;
    __shared__ float red[256];

    float vals[3];
    float sum = 0.f;
    #pragma unroll
    for (int e = 0; e < 3; ++e) {
        const int h = e * 256 + tid;
        const long long idx = (long long)tok * H + h;
        const float v = x[idx] + y[idx];
        vals[e] = v; sum += v;
    }
    red[tid] = sum; __syncthreads();
    for (int off = 128; off > 0; off >>= 1) {
        if (tid < off) red[tid] += red[tid + off];
        __syncthreads();
    }
    const float mean = red[0] * (1.0f / H);
    __syncthreads();
    float ss = 0.f;
    #pragma unroll
    for (int e = 0; e < 3; ++e) { const float d = vals[e] - mean; ss += d * d; }
    red[tid] = ss; __syncthreads();
    for (int off = 128; off > 0; off >>= 1) {
        if (tid < off) red[tid] += red[tid + off];
        __syncthreads();
    }
    const float rstd = rsqrtf(red[0] * (1.0f / H) + 1e-12f);
    #pragma unroll
    for (int e = 0; e < 3; ++e) {
        const int h = e * 256 + tid;
        const long long idx = (long long)tok * H + h;
        const float v = (vals[e] - mean) * rstd * g[h] + bta[h];
        x[idx]  = v;
        xb[idx] = (bf16_t)v;
    }
}

// ---------------------------------------------------------------------------
// Softmax over one score row (scale + additive mask), writes bf16 probs.
// blockIdx.x = (b*NH+h)*SEQ + q. 128 threads, 4 elems each.
// ---------------------------------------------------------------------------
__global__ __launch_bounds__(128) void softmax_row(
    const float* __restrict__ scores, const int* __restrict__ amask,
    bf16_t* __restrict__ probs, float scale)
{
    const int row = blockIdx.x;
    const int zz  = row >> 9;       // b*NH + h
    const int b   = zz / NH;
    const float* sp = scores + (long long)row * SEQ;
    bf16_t* pp      = probs  + (long long)row * SEQ;
    const int* mk   = amask + b * SEQ;
    const int tid = threadIdx.x;
    __shared__ float red[128];

    float v[4];
    float mx = -3.0e38f;
    #pragma unroll
    for (int e = 0; e < 4; ++e) {
        const int t = e * 128 + tid;
        const float s = sp[t] * scale + (mk[t] ? 0.0f : -1.0e9f);
        v[e] = s; mx = fmaxf(mx, s);
    }
    red[tid] = mx; __syncthreads();
    for (int off = 64; off > 0; off >>= 1) {
        if (tid < off) red[tid] = fmaxf(red[tid], red[tid + off]);
        __syncthreads();
    }
    mx = red[0]; __syncthreads();
    float sum = 0.f;
    #pragma unroll
    for (int e = 0; e < 4; ++e) { v[e] = expf(v[e] - mx); sum += v[e]; }
    red[tid] = sum; __syncthreads();
    for (int off = 64; off > 0; off >>= 1) {
        if (tid < off) red[tid] += red[tid + off];
        __syncthreads();
    }
    const float inv = 1.0f / red[0];
    #pragma unroll
    for (int e = 0; e < 4; ++e) pp[e * 128 + tid] = (bf16_t)(v[e] * inv);
}

// ---------------------------------------------------------------------------
// Vt[b][h][d][s] = qkv_bf[(b*SEQ+s)*3H + 2H + h*DH + d]
// ---------------------------------------------------------------------------
__global__ void make_vt(const bf16_t* __restrict__ qkvb, bf16_t* __restrict__ vt)
{
    long long idx = (long long)blockIdx.x * 256 + threadIdx.x;
    const int s = (int)(idx & (SEQ - 1));
    long long r = idx >> 9;
    const int d = (int)(r & (DH - 1)); r >>= 6;
    const int h = (int)(r % NH);
    const int b = (int)(r / NH);
    vt[idx] = qkvb[((long long)(b * SEQ + s)) * (3 * H) + 2 * H + h * DH + d];
}

// W[K,N] f32 -> Wt[N,K] bf16
__global__ void wcvt_t(const float* __restrict__ W, bf16_t* __restrict__ Wt,
                       int K, int N)
{
    const long long idx = (long long)blockIdx.x * 256 + threadIdx.x;
    if (idx >= (long long)K * N) return;
    const int k = (int)(idx / N);
    const int n = (int)(idx - (long long)k * N);
    Wt[(long long)n * K + k] = (bf16_t)W[idx];
}

__global__ void cvt_bf16(const float* __restrict__ s, bf16_t* __restrict__ d,
                         long long n)
{
    const long long idx = (long long)blockIdx.x * 256 + threadIdx.x;
    if (idx < n) d[idx] = (bf16_t)s[idx];
}

// dst[row] = sum_c src[row,c]^2
__global__ __launch_bounds__(256) void rowsumsq(const float* __restrict__ src,
                                                float* __restrict__ dst, int cols)
{
    const int row = blockIdx.x, tid = threadIdx.x;
    __shared__ float red[256];
    float s = 0.f;
    for (int c = tid; c < cols; c += 256) {
        const float v = src[(long long)row * cols + c];
        s += v * v;
    }
    red[tid] = s; __syncthreads();
    for (int off = 128; off > 0; off >>= 1) {
        if (tid < off) red[tid] += red[tid + off];
        __syncthreads();
    }
    if (tid == 0) dst[row] = red[0];
}

__global__ void init2(float* p) { if (threadIdx.x < 2) p[threadIdx.x] = 0.0f; }

// gram[row,t] -> dist (in place), accumulate global sum. row = b*SEQ+s.
__global__ __launch_bounds__(256) void dist_ww(float* __restrict__ gram,
                                               const float* __restrict__ sq,
                                               float* __restrict__ dsum)
{
    const int row = blockIdx.x, tid = threadIdx.x;
    const int b   = row >> 9;
    float* gp = gram + (long long)row * SEQ;
    const float sqa = sq[row];
    __shared__ float red[256];
    float s = 0.f;
    for (int t = tid; t < SEQ; t += 256) {
        const float g = gp[t];
        const float d = sqrtf(fmaxf(sqa + sq[b * SEQ + t] - 2.0f * g, 0.0f));
        gp[t] = d;
        s += d;
    }
    red[tid] = s; __syncthreads();
    for (int off = 128; off > 0; off >>= 1) {
        if (tid < off) red[tid] += red[tid + off];
        __syncthreads();
    }
    if (tid == 0) atomicAdd(dsum, red[0]);
}

__global__ void fin_ww(const float* __restrict__ dist,
                       const float* __restrict__ dsum, float* __restrict__ out)
{
    const long long idx = (long long)blockIdx.x * 256 + threadIdx.x;
    const float mean = dsum[0] * (1.0f / ((float)BB * SEQ * SEQ));
    const float v = 1.0f / (1.0f + expf(dist[idx] - mean)); // sigmoid(mean-dist)
    out[idx] = (v < 0.4f) ? 0.0f : v;
}

__global__ __launch_bounds__(32) void dist_wo(float* __restrict__ disto,
                                              const float* __restrict__ sq,
                                              const float* __restrict__ sqo,
                                              float* __restrict__ dsumo)
{
    const int row = blockIdx.x, tid = threadIdx.x;
    __shared__ float red[32];
    float v = 0.f;
    if (tid < NOPS) {
        const float g = disto[(long long)row * NOPS + tid];
        const float d = sqrtf(fmaxf(sq[row] + sqo[tid] - 2.0f * g, 0.0f));
        disto[(long long)row * NOPS + tid] = d;
        v = d;
    }
    red[tid] = v; __syncthreads();
    for (int off = 16; off > 0; off >>= 1) {
        if (tid < off) red[tid] += red[tid + off];
        __syncthreads();
    }
    if (tid == 0) atomicAdd(dsumo, red[0]);
}

__global__ void fin_wo(const float* __restrict__ disto,
                       const float* __restrict__ dsum, float* __restrict__ out)
{
    const long long idx = (long long)blockIdx.x * 256 + threadIdx.x;
    const float mean = dsum[1] * (1.0f / ((float)BB * SEQ * NOPS));
    const float v = 1.0f / (1.0f + expf(disto[idx] - mean));
    out[idx] = (v < 0.4f) ? 0.0f : v;
}

// ---------------------------------------------------------------------------
extern "C" void kernel_launch(void* const* d_in, const int* in_sizes, int n_in,
                              void* d_out, int out_size, void* d_ws,
                              size_t ws_size, hipStream_t stream)
{
    (void)in_sizes; (void)n_in; (void)out_size; (void)ws_size;

    const long long* ids = (const long long*)d_in[0];
    const int* amask     = (const int*)d_in[1];
    const float* wemb    = (const float*)d_in[2];
    const float* pemb    = (const float*)d_in[3];
    const float* temb    = (const float*)d_in[4];
    const float* eg      = (const float*)d_in[5];
    const float* ebt     = (const float*)d_in[6];
    const float* Wqkv    = (const float*)d_in[7];
    const float* bqkv    = (const float*)d_in[8];
    const float* Wo      = (const float*)d_in[9];
    const float* bo      = (const float*)d_in[10];
    const float* ln1g    = (const float*)d_in[11];
    const float* ln1b    = (const float*)d_in[12];
    const float* W1      = (const float*)d_in[13];
    const float* b1      = (const float*)d_in[14];
    const float* W2      = (const float*)d_in[15];
    const float* b2      = (const float*)d_in[16];
    const float* ln2g    = (const float*)d_in[17];
    const float* ln2b    = (const float*)d_in[18];
    const float* opemb   = (const float*)d_in[19];

    char* wp = (char*)d_ws;
    auto alloc = [&](size_t bytes) -> void* {
        void* p = (void*)wp;
        wp += (bytes + 255) & ~(size_t)255;
        return p;
    };

    float*  x      = (float*) alloc((size_t)NTOK * H * 4);
    bf16_t* xb     = (bf16_t*)alloc((size_t)NTOK * H * 2);
    bf16_t* qkvb   = (bf16_t*)alloc((size_t)NTOK * 3 * H * 2);
    float*  scores = (float*) alloc((size_t)BB * NH * SEQ * SEQ * 4);
    bf16_t* probs  = (bf16_t*)alloc((size_t)BB * NH * SEQ * SEQ * 2);
    bf16_t* vt     = (bf16_t*)alloc((size_t)BB * NH * DH * SEQ * 2);
    bf16_t* ctxb   = (bf16_t*)alloc((size_t)NTOK * H * 2);
    float*  y      = (float*) alloc((size_t)NTOK * H * 4);
    bf16_t* hb     = (bf16_t*)alloc((size_t)NTOK * FF * 2);
    bf16_t* WqkvT  = (bf16_t*)alloc((size_t)H * 3 * H * 2);
    bf16_t* WoT    = (bf16_t*)alloc((size_t)H * H * 2);
    bf16_t* W1T    = (bf16_t*)alloc((size_t)H * FF * 2);
    bf16_t* W2T    = (bf16_t*)alloc((size_t)FF * H * 2);
    float*  sq     = (float*) alloc((size_t)NTOK * 4);
    float*  sqo    = (float*) alloc((size_t)NOPS * 4);
    bf16_t* opb    = (bf16_t*)alloc((size_t)NOPS * H * 2);
    float*  disto  = (float*) alloc((size_t)NTOK * NOPS * 4);
    float*  dsum   = (float*) alloc(2 * 4);

    float* out_emb = (float*)d_out;
    float* out_ww  = out_emb + (size_t)NTOK * H;
    float* out_wo  = out_ww + (size_t)BB * SEQ * SEQ;

    const dim3 blk(256);

    embed_ln<<<NTOK, 256, 0, stream>>>(ids, wemb, pemb, temb, eg, ebt, x, xb);

    const float scale = 0.125f; // 1/sqrt(DH)

    for (int li = 0; li < NL; ++li) {
        const float* Wqkv_l = Wqkv + (size_t)li * H * 3 * H;
        const float* Wo_l   = Wo   + (size_t)li * H * H;
        const float* W1_l   = W1   + (size_t)li * H * FF;
        const float* W2_l   = W2   + (size_t)li * FF * H;

        wcvt_t<<<(H * 3 * H + 255) / 256, 256, 0, stream>>>(Wqkv_l, WqkvT, H, 3 * H);
        wcvt_t<<<(H * H + 255) / 256, 256, 0, stream>>>(Wo_l, WoT, H, H);
        wcvt_t<<<(H * FF + 255) / 256, 256, 0, stream>>>(W1_l, W1T, H, FF);
        wcvt_t<<<(FF * H + 255) / 256, 256, 0, stream>>>(W2_l, W2T, FF, H);

        // qkv (bf16) = xb @ Wqkv + bqkv
        gemm_bf16_tn<<<dim3(3 * H / 128, NTOK / 64, 1), blk, 0, stream>>>(
            xb, H, 0, 0, WqkvT, H, 0, 0,
            nullptr, qkvb, 3 * H, 0, 0,
            bqkv + (size_t)li * 3 * H, NTOK, 3 * H, H, 1, 0);

        // scores[b,h] = Q @ K^T   (TN directly on the qkv buffer)
        gemm_bf16_tn<<<dim3(SEQ / 128, SEQ / 64, BB * NH), blk, 0, stream>>>(
            qkvb,     3 * H, (long long)SEQ * 3 * H, DH,
            qkvb + H, 3 * H, (long long)SEQ * 3 * H, DH,
            scores, nullptr, SEQ,
            (long long)NH * SEQ * SEQ, (long long)SEQ * SEQ,
            nullptr, SEQ, SEQ, DH, NH, 0);

        softmax_row<<<BB * NH * SEQ, 128, 0, stream>>>(scores, amask, probs, scale);

        make_vt<<<(BB * NH * DH * SEQ) / 256, 256, 0, stream>>>(qkvb, vt);

        // ctx[b,h] = probs @ V   (Bt = V^T)
        gemm_bf16_tn<<<dim3(1, SEQ / 64, BB * NH), blk, 0, stream>>>(
            probs, SEQ, (long long)NH * SEQ * SEQ, (long long)SEQ * SEQ,
            vt,    SEQ, (long long)NH * DH * SEQ,  (long long)DH * SEQ,
            nullptr, ctxb, H, (long long)SEQ * H, DH,
            nullptr, SEQ, DH, SEQ, NH, 0);

        // y = ctx @ Wo + bo
        gemm_bf16_tn<<<dim3(H / 128, NTOK / 64, 1), blk, 0, stream>>>(
            ctxb, H, 0, 0, WoT, H, 0, 0,
            y, nullptr, H, 0, 0,
            bo + (size_t)li * H, NTOK, H, H, 1, 0);

        ln_residual<<<NTOK, 256, 0, stream>>>(
            x, y, ln1g + (size_t)li * H, ln1b + (size_t)li * H, xb);

        // h (bf16) = gelu(x @ W1 + b1)
        gemm_bf16_tn<<<dim3(FF / 128, NTOK / 64, 1), blk, 0, stream>>>(
            xb, H, 0, 0, W1T, H, 0, 0,
            nullptr, hb, FF, 0, 0,
            b1 + (size_t)li * FF, NTOK, FF, H, 1, 1);

        // y = h @ W2 + b2
        gemm_bf16_tn<<<dim3(H / 128, NTOK / 64, 1), blk, 0, stream>>>(
            hb, FF, 0, 0, W2T, FF, 0, 0,
            y, nullptr, H, 0, 0,
            b2 + (size_t)li * H, NTOK, H, FF, 1, 0);

        ln_residual<<<NTOK, 256, 0, stream>>>(
            x, y, ln2g + (size_t)li * H, ln2b + (size_t)li * H, xb);
    }

    // ---- outputs ----
    hipMemcpyAsync(out_emb, x, (size_t)NTOK * H * 4,
                   hipMemcpyDeviceToDevice, stream);

    init2<<<1, 32, 0, stream>>>(dsum);
    rowsumsq<<<NTOK, 256, 0, stream>>>(x, sq, H);

    // gram[b] = emb @ emb^T (reuse scores buffer)
    gemm_bf16_tn<<<dim3(SEQ / 128, SEQ / 64, BB), blk, 0, stream>>>(
        xb, H, (long long)SEQ * H, 0,
        xb, H, (long long)SEQ * H, 0,
        scores, nullptr, SEQ, (long long)SEQ * SEQ, 0,
        nullptr, SEQ, SEQ, H, 1, 0);

    dist_ww<<<NTOK, 256, 0, stream>>>(scores, sq, dsum);
    fin_ww<<<(BB * SEQ * SEQ) / 256, 256, 0, stream>>>(scores, dsum, out_ww);

    cvt_bf16<<<(NOPS * H + 255) / 256, 256, 0, stream>>>(opemb, opb, NOPS * H);
    rowsumsq<<<NOPS, 256, 0, stream>>>(opemb, sqo, H);

    // gramo = emb @ op_emb^T
    gemm_bf16_tn<<<dim3(1, NTOK / 64, 1), blk, 0, stream>>>(
        xb, H, 0, 0, opb, H, 0, 0,
        disto, nullptr, NOPS, 0, 0,
        nullptr, NTOK, NOPS, H, 1, 0);

    dist_wo<<<NTOK, 32, 0, stream>>>(disto, sq, sqo, dsum);
    fin_wo<<<(NTOK * NOPS) / 256, 256, 0, stream>>>(disto, dsum, out_wo);
}
